// ARSLIFBiLSTM_79955111182812
// MI455X (gfx1250) — matrix operand order repaired and checked by reading, not compiled, not run on hardware
//
#include <hip/hip_runtime.h>

typedef __attribute__((ext_vector_type(16))) _Float16 v16h;
typedef __attribute__((ext_vector_type(8)))  _Float16 v8h;
typedef __attribute__((ext_vector_type(8)))  float    v8f;

#define T_SEQ 1024
#define DIM   256       // INPUT_DIM
#define UNITS 256
#define G4    1024      // 4*UNITS
#define KTOT  512       // DIM + UNITS (fused [x|h] K dimension)
#define BT    16        // batch-tile rows (WMMA M)

// LDS row strides (padded to dodge bank conflicts)
#define ZS 1040         // z/spike/Vm/Vth row stride in floats
#define XS 520          // xh row stride in halves

// ---------------------------------------------------------------------------
// Phase 0: pack fused [W;U] into *fragment-major* f16 layout:
//   CwTf[d][frag][lane][hh],  frag = nt*16 + kt  (nt: N-tile 0..63, kt: K-tile 0..15)
// Lane l of a fragment holds column n = nt*16 + (l&15),
// K halves k = kt*32 + ((l&16)?16:0) + hh  (contiguous k order; A uses the same).
// A wave's B-fragment load is then 32 lanes x 32B = 1KB fully contiguous.
// Total 2 MB -> L2 resident.
// ---------------------------------------------------------------------------
__global__ void pack_weights(const float* __restrict__ Wf, const float* __restrict__ Uf,
                             const float* __restrict__ Wb, const float* __restrict__ Ub,
                             _Float16* __restrict__ CwTf) {
    int idx = blockIdx.x * blockDim.x + threadIdx.x;   // 0 .. 2*2^19-1
    int hh   = idx & 15;
    int lane = (idx >> 4) & 31;
    int kt   = (idx >> 9) & 15;
    int nt   = (idx >> 13) & 63;
    int d    = idx >> 19;
    int n = nt * 16 + (lane & 15);
    int k = kt * 32 + ((lane & 16) ? 16 : 0) + hh;
    const float* W = d ? Wb : Wf;
    const float* U = d ? Ub : Uf;
    float v = (k < DIM) ? W[k * G4 + n] : U[(k - DIM) * G4 + n];
    CwTf[idx] = (_Float16)v;
}

// ---------------------------------------------------------------------------
// Phase 1: persistent recurrent kernel. One WG = (direction, 16-row batch
// tile). 512 threads = 16 wave32 waves. Per step:
//   z[16,1024] = [x_t | h][16,512] @ Cw^T    (v_wmma_f32_16x16x32_f16)
//   LIF update + row reductions + threshold adapt + LSTM gating, all in LDS.
// B fragments stream from the 2MB L2-resident table each step. The K-tile
// order is rotated by (t & 15) so the 64 fragment addresses are t-dependent:
// this defeats LICM (no hoist-and-spill, no volatile/SYS-scope loads) while
// keeping plain cached global_load_b128 in the loop. One-group software
// pipeline: group kt+1 is issued before the WMMAs of group kt consume data.
// ---------------------------------------------------------------------------
__launch_bounds__(512)
__global__ void arslif_recurrent(const float* __restrict__ x,
                                 const _Float16* __restrict__ CwTf_all,
                                 const float* __restrict__ bf, const float* __restrict__ Rf,
                                 const float* __restrict__ bb, const float* __restrict__ Rb,
                                 float* __restrict__ out) {
    extern __shared__ char smem[];
    float*    zs     = (float*)smem;                 // [16][ZS]  z, then spikes
    float*    vm     = zs  + BT * ZS;                // [16][ZS]  membrane
    float*    vth    = vm  + BT * ZS;                // [16][ZS]  threshold
    _Float16* xh     = (_Float16*)(vth + BT * ZS);   // [16][XS]  fused [x_t|h] f16
    float*    Rsh    = (float*)(xh + BT * XS);       // [1024]
    float*    bsh    = Rsh + G4;                     // [1024]
    float*    reds   = bsh + G4;                     // [16][32] spike partials
    float*    redf   = reds + BT * 32;               // [16][32] f_adapt partials
    float*    rowAdj = redf + BT * 32;               // [16]

    const int tid  = threadIdx.x;
    const int lane = tid & 31;
    const int wave = tid >> 5;
    const int d    = blockIdx.x >> 2;                // direction
    const int bt   = blockIdx.x & 3;                 // batch tile
    const int b0   = bt * BT;

    // fragment-major B table: Bv[f*32] (lane folded in) = this lane's 32B of frag f
    const v16h* __restrict__ Bv =
        (const v16h*)(CwTf_all + ((size_t)d << 19)) + lane;
    const float* bvec = d ? bb : bf;
    const float* Rvec = d ? Rb : Rf;

    // ---- init ----
    Rsh[tid] = Rvec[tid];  Rsh[tid + 512] = Rvec[tid + 512];
    bsh[tid] = bvec[tid];  bsh[tid + 512] = bvec[tid + 512];

    const int arow = tid >> 5;            // 0..15 (elementwise row)
    const int ac0  = (tid & 31) * 32;     // 32 cols per thread
    const int gu0  = (tid & 31) * 8;      // 8 units per thread (gating)

    #pragma unroll
    for (int j = 0; j < 32; ++j) {
        vm [arow * ZS + ac0 + j] = 0.f;
        vth[arow * ZS + ac0 + j] = 0.f;
    }
    #pragma unroll
    for (int j = 0; j < 8; ++j) xh[arow * XS + DIM + gu0 + j] = (_Float16)0.f;

    float creg[8];
    #pragma unroll
    for (int j = 0; j < 8; ++j) creg[j] = 0.f;

    // WMMA per-lane fragment geometry (wave32 layouts, CDNA5 ISA §7.12.2)
    const int am   = lane & 15;                 // A: M row
    const int aklo = (lane & 16) ? 16 : 0;      // A: K chunk base
    const int nc   = lane & 15;                 // C: N within tile
    const int rb   = (lane & 16) ? 8 : 0;       // C: M base

    // per-wave fragment bases: frag(nt,kt) = nt*16 + kt, nt in {w, w+16, w+32, w+48}
    const int fb0 = (wave     ) * 16;
    const int fb1 = (wave + 16) * 16;
    const int fb2 = (wave + 32) * 16;
    const int fb3 = (wave + 48) * 16;

    for (int t = 0; t < T_SEQ; ++t) {
        const int tg  = d ? (T_SEQ - 1 - t) : t;  // bwd reads reversed input
        const int rot = t & 15;                   // t-dependent K-tile rotation

        // ---- stage x_t tile into xh (f32 -> f16) ----
        {
            const float* xp = x + ((size_t)(b0 + arow) * T_SEQ + tg) * DIM + lane * 8;
            #pragma unroll
            for (int j = 0; j < 8; ++j)
                xh[arow * XS + lane * 8 + j] = (_Float16)xp[j];
        }
        __syncthreads();

        // ---- z = [x|h] @ Cw^T, software-pipelined, rotated K order ----
        {
            v8f acc0 = {}, acc1 = {}, acc2 = {}, acc3 = {};
            v16h bc0 = Bv[(size_t)(fb0 + rot) * 32];
            v16h bc1 = Bv[(size_t)(fb1 + rot) * 32];
            v16h bc2 = Bv[(size_t)(fb2 + rot) * 32];
            v16h bc3 = Bv[(size_t)(fb3 + rot) * 32];
            #pragma unroll
            for (int kt = 0; kt < 16; ++kt) {
                const int ke = (kt + rot) & 15;        // current K-tile
                v16h bn0, bn1, bn2, bn3;
                if (kt < 15) {                         // prefetch next K-group
                    const int kn = (kt + 1 + rot) & 15;
                    bn0 = Bv[(size_t)(fb0 + kn) * 32];
                    bn1 = Bv[(size_t)(fb1 + kn) * 32];
                    bn2 = Bv[(size_t)(fb2 + kn) * 32];
                    bn3 = Bv[(size_t)(fb3 + kn) * 32];
                }
                const _Float16* ap = xh + am * XS + ke * 32 + aklo;
                v8h lo = *(const v8h*)ap;
                v8h hi = *(const v8h*)(ap + 8);
                v16h a = __builtin_shufflevector(lo, hi,
                         0,1,2,3,4,5,6,7,8,9,10,11,12,13,14,15);
                acc0 = __builtin_amdgcn_wmma_f32_16x16x32_f16(false, a, false, bc0, (short)0, acc0, false, false);
                acc1 = __builtin_amdgcn_wmma_f32_16x16x32_f16(false, a, false, bc1, (short)0, acc1, false, false);
                acc2 = __builtin_amdgcn_wmma_f32_16x16x32_f16(false, a, false, bc2, (short)0, acc2, false, false);
                acc3 = __builtin_amdgcn_wmma_f32_16x16x32_f16(false, a, false, bc3, (short)0, acc3, false, false);
                if (kt < 15) { bc0 = bn0; bc1 = bn1; bc2 = bn2; bc3 = bn3; }
            }
            #pragma unroll
            for (int r = 0; r < 8; ++r) {
                zs[(r + rb) * ZS + (wave     ) * 16 + nc] = acc0[r];
                zs[(r + rb) * ZS + (wave + 16) * 16 + nc] = acc1[r];
                zs[(r + rb) * ZS + (wave + 32) * 16 + nc] = acc2[r];
                zs[(r + rb) * ZS + (wave + 48) * 16 + nc] = acc3[r];
            }
        }
        __syncthreads();

        // ---- LIF membrane + surrogate spikes + per-row partial sums ----
        {
            float sspk = 0.f, sfad = 0.f;
            #pragma unroll
            for (int j = 0; j < 32; ++j) {
                const int col = ac0 + j;
                const int o   = arow * ZS + col;
                float z   = zs[o] + bsh[col];
                float rz  = Rsh[col] * z;
                float v   = vm[o] * 0.9f + 0.1f * rz;          // DT/TAU_MEM = 0.1
                float th  = vth[o];
                float vsc = v / (th + 1e-7f);
                float spk = 1.f / (1.f + __expf(1.0f - vsc));  // sigmoid(vsc - 1)
                vm[o] = (spk > 0.f) ? 0.f : v;                 // spike reset
                sspk += spk;
                sfad += (fabsf(rz) >= th) ? 0.99f : 0.01f;     // F_ACTIVE / F_REST
                zs[o] = spk;                                   // overwrite z with spikes
            }
            reds[arow * 32 + (tid & 31)] = sspk;
            redf[arow * 32 + (tid & 31)] = sfad;
        }
        __syncthreads();

        // ---- adaptive threshold rate (one thread per batch row) ----
        if (tid < BT) {
            float ss = 0.f, sf = 0.f;
            #pragma unroll
            for (int j = 0; j < 32; ++j) { ss += reds[tid * 32 + j]; sf += redf[tid * 32 + j]; }
            float r  = fminf(fmaxf(ss * (1.f / 1024.f), 0.f), 1.f);
            float mf = sf * (1.f / 1024.f);
            float re = 0.6f * (r - 0.65f) + 0.4f * (r - mf);   // ALPHA/BETA/F
            rowAdj[tid] = 1e-3f * re;                          // DT/TAU_ADAPT
        }
        __syncthreads();

        // ---- Vth update + LSTM gating on spikes ----
        {
            const float adj = rowAdj[arow];
            #pragma unroll
            for (int j = 0; j < 32; ++j) vth[arow * ZS + ac0 + j] += adj;

            float* orow = out + ((size_t)(b0 + arow) * T_SEQ + t) * (2 * UNITS)
                              + d * UNITS + gu0;
            #pragma unroll
            for (int j = 0; j < 8; ++j) {
                const int u = gu0 + j;
                float s0 = zs[arow * ZS +             u];
                float s1 = zs[arow * ZS +     UNITS + u];
                float s2 = zs[arow * ZS + 2 * UNITS + u];
                float s3 = zs[arow * ZS + 3 * UNITS + u];
                float ig = 1.f / (1.f + __expf(-s0));
                float fg = 1.f / (1.f + __expf(-s1));
                float gg = tanhf(s2);
                float og = 1.f / (1.f + __expf(-s3));
                float c  = fg * creg[j] + ig * gg;
                creg[j]  = c;
                float h  = og * tanhf(c);
                xh[arow * XS + DIM + u] = (_Float16)h;         // feed next step's A
                orow[j] = h;
            }
        }
        __syncthreads();
    }
}

// ---------------------------------------------------------------------------
extern "C" void kernel_launch(void* const* d_in, const int* in_sizes, int n_in,
                              void* d_out, int out_size, void* d_ws, size_t ws_size,
                              hipStream_t stream) {
    (void)in_sizes; (void)n_in; (void)out_size; (void)ws_size;
    const float* x  = (const float*)d_in[0];
    const float* Wf = (const float*)d_in[1];
    const float* Uf = (const float*)d_in[2];
    const float* bf = (const float*)d_in[3];
    const float* Rf = (const float*)d_in[4];
    const float* Wb = (const float*)d_in[5];
    const float* Ub = (const float*)d_in[6];
    const float* bb = (const float*)d_in[7];
    const float* Rb = (const float*)d_in[8];

    _Float16* CwTf = (_Float16*)d_ws;   // 2 MB packed fragment-major weights

    pack_weights<<<4096, 256, 0, stream>>>(Wf, Uf, Wb, Ub, CwTf);

    // LDS: 3*16*1040*4 + 16*520*2 + 2*4096 + 2*2048 + 64 = 228672 bytes
    const size_t lds = (size_t)(3 * BT * ZS * 4) + (size_t)(BT * XS * 2)
                     + 2 * G4 * 4 + 2 * BT * 32 * 4 + BT * 4;
    arslif_recurrent<<<8, 512, lds, stream>>>(x, CwTf, bf, Rf, bb, Rb, (float*)d_out);
}
